// InterModalityCorrelation_30073361007270
// MI455X (gfx1250) — matrix operand order repaired and checked by reading, not compiled, not run on hardware
//
#include <hip/hip_runtime.h>

typedef _Float16 v8h  __attribute__((ext_vector_type(8)));
typedef _Float16 v16h __attribute__((ext_vector_type(16)));
typedef float    v4f  __attribute__((ext_vector_type(4)));
typedef float    v8f  __attribute__((ext_vector_type(8)));

#define WMMA_F32_F16(a, b, c) \
  __builtin_amdgcn_wmma_f32_16x16x32_f16(false, (a), false, (b), (short)0, (c), false, false)

static __device__ __forceinline__ v16h cat16(v8h lo, v8h hi) {
  return __builtin_shufflevector(lo, hi, 0,1,2,3,4,5,6,7,8,9,10,11,12,13,14,15);
}

// ---------------------------------------------------------------------------
// Weight conversion: fp32 -> f16, K-major "B operand" layout dst[n*256 + k].
// ---------------------------------------------------------------------------
__global__ __launch_bounds__(256) void conv_qkv_w(const float* __restrict__ src,
                                                  _Float16* __restrict__ dst) {
  // src shape (H=8, K=256, E=32), column n = h*32 + e
  int i = blockIdx.x * 256 + threadIdx.x;      // 65536 total
  int e = i & 31, k = (i >> 5) & 255, h = i >> 13;
  dst[(h * 32 + e) * 256 + k] = (_Float16)src[i];
}

__global__ __launch_bounds__(256) void conv_sq_w(const float* __restrict__ src,
                                                 _Float16* __restrict__ dst) {
  // src shape (K=256, N=256) row-major
  int i = blockIdx.x * 256 + threadIdx.x;      // 65536 total
  int n = i & 255, k = i >> 8;
  dst[n * 256 + k] = (_Float16)src[i];
}

// ---------------------------------------------------------------------------
// Generic 16-row x 256-col projection:  Y = X(16x256) @ W(256x256) + bias
// 8 waves per block, each wave does two 16x16 output tiles, K looped by 32.
// f16 input path uses async DMA global->LDS (ASYNCcnt). Outputs staged in LDS
// and written as packed coalesced stores.
// mode 0: f16 row-major (ld 256) | mode 1: f16 transposed (ld ldT) | mode 2: f32
// pad/Srows implement the zero-padded kv sequence (padded rows = bias only).
// ---------------------------------------------------------------------------
__global__ __launch_bounds__(256) void proj16(
    const void* __restrict__ Xv, int xIsF16, long xBatch,
    const _Float16* __restrict__ W, const float* __restrict__ bias,
    void* __restrict__ Yv, int mode, long yBatch, int ldT,
    int Srows, int pad) {
  __shared__ __align__(32) char smem[16 * 264 * 4];  // aliased: X tile, then Y tile
  _Float16* Xs = (_Float16*)smem;                    // f16 stride 272 halves (544B)
  const int tid = threadIdx.x;
  const int rowbase = blockIdx.x * 16;
  const int by = blockIdx.y;

  if (xIsF16) {
    // Async DMA 16 rows x 512B into LDS (caller guarantees rows in-range, pad=0).
    const char* src = (const char*)Xv + (xBatch * by + (long)rowbase * 256) * 2;
    #pragma unroll
    for (int i = 0; i < 2; ++i) {
      int unit = tid + 256 * i;                      // 512 units: 16 rows x 32 chunks
      int row = unit >> 5, ch = unit & 31;
      unsigned lds = (unsigned)(size_t)smem + row * 544 + ch * 16;
      unsigned long long ga = (unsigned long long)(src + (long)row * 512 + ch * 16);
      asm volatile("global_load_async_to_lds_b128 %0, %1, off"
                   :: "v"(lds), "v"(ga) : "memory");
    }
    asm volatile("s_wait_asynccnt 0" ::: "memory");
  } else {
    // fp32 load + convert (handles zero padding of kv sequence)
    for (int r = 0; r < 16; ++r) {
      int sr = rowbase + r - pad;
      float val = 0.f;
      if (sr >= 0 && sr < Srows) {
        const float* xp = (const float*)Xv + xBatch * by + (long)sr * 256 + tid;
        val = *xp;
        if (r == 0) __builtin_prefetch(xp + 16 * 256, 0, 1);  // next row tile
      }
      Xs[r * 272 + tid] = (_Float16)val;
    }
  }
  __syncthreads();

  const int w  = tid >> 5;
  const int ln = tid & 31;
  const int hl = ln >> 4;                            // half-wave selector
  const int m  = ln & 15;

  v8f acc0 = {}, acc1 = {};
  const int n0a = w * 16, n0b = 128 + w * 16;

  #pragma unroll
  for (int kc = 0; kc < 8; ++kc) {
    const int k0 = kc * 32;
    // A operand (16x32 f16): lane row = m, K chunks per half-wave
    v8h alo = *(const v8h*)&Xs[m * 272 + k0 + hl * 16];
    v8h ahi = *(const v8h*)&Xs[m * 272 + k0 + hl * 16 + 8];
    v16h a = cat16(alo, ahi);
    // B operand (32x16 f16): lane col = m, 16 contiguous K halves
    v16h b0 = *(const v16h*)&W[(n0a + m) * 256 + k0 + hl * 16];
    v16h b1 = *(const v16h*)&W[(n0b + m) * 256 + k0 + hl * 16];
    acc0 = WMMA_F32_F16(a, b0, acc0);
    acc1 = WMMA_F32_F16(a, b1, acc1);
  }
  __syncthreads();                                   // done reading Xs; reuse smem

  #pragma unroll
  for (int t = 0; t < 2; ++t) {
    const v8f acc = t ? acc1 : acc0;
    const int col = (t ? n0b : n0a) + m;
    const float bb = bias[col];
    if (mode == 1) {
      // transposed f16: rows contiguous per lane -> packed 16B store (no staging)
      v8h pk;
      #pragma unroll
      for (int v = 0; v < 8; ++v) pk[v] = (_Float16)(acc[v] + bb);
      long base = yBatch * by + (long)col * ldT + rowbase + hl * 8;
      *(v8h*)((_Float16*)Yv + base) = pk;
    } else if (mode == 0) {
      _Float16* Ys = (_Float16*)smem;                // stride 272
      #pragma unroll
      for (int v = 0; v < 8; ++v)
        Ys[(v + hl * 8) * 272 + col] = (_Float16)(acc[v] + bb);
    } else {
      float* Yf = (float*)smem;                      // stride 264
      #pragma unroll
      for (int v = 0; v < 8; ++v)
        Yf[(v + hl * 8) * 264 + col] = acc[v] + bb;
    }
  }
  __syncthreads();

  // Packed, coalesced tile store from LDS
  if (mode == 0) {
    int row = tid >> 4, ch = tid & 15;               // 16 chunks of 16 halves
    const _Float16* Ys = (const _Float16*)smem;
    _Float16* dst = (_Float16*)Yv + yBatch * by + (long)(rowbase + row) * 256 + ch * 16;
    *(v16h*)dst = *(const v16h*)&Ys[row * 272 + ch * 16];
  } else if (mode == 2) {
    int row = tid >> 4, ch = tid & 15;               // 16 chunks of 16 floats
    const float* Yf = (const float*)smem;
    float* dst = (float*)Yv + yBatch * by + (long)(rowbase + row) * 256 + ch * 16;
    #pragma unroll
    for (int j = 0; j < 4; ++j)
      *(v4f*)(dst + j * 4) = *(const v4f*)&Yf[row * 264 + ch * 16 + j * 4];
  }
}

// ---------------------------------------------------------------------------
// Windowed attention (WINDOW=31, HEAD_DIM=32). One wave per 16-query tile.
// Scores: 3 WMMAs over 48 key slots; masked softmax (shfl within 16-lane half);
// P reshaped via LDS; out = P x V via 4 WMMAs; output staged -> packed stores.
// ---------------------------------------------------------------------------
__global__ __launch_bounds__(128) void attn31(
    const _Float16* __restrict__ Q,    // [B*S][256]   (head-major cols)
    const _Float16* __restrict__ K,    // [B][2080][256]
    const _Float16* __restrict__ Vt,   // [B][256][2112] (dim-major, key contiguous)
    _Float16* __restrict__ O) {        // [B*S][256]
  __shared__ _Float16 P[4][16][48];
  __shared__ __align__(32) _Float16 Ot[4][16][32];
  const int w  = threadIdx.x >> 5;
  const int ln = threadIdx.x & 31;
  const int hl = ln >> 4;
  const int m  = ln & 15;
  const int h  = blockIdx.y, b = blockIdx.z;
  const int s0 = (blockIdx.x * 4 + w) * 16;

  // Q tile as A operand (16 queries x 32 dims)
  const _Float16* qb = Q + ((long)(b * 2048 + s0 + m)) * 256 + h * 32;
  v16h aq = cat16(*(const v8h*)(qb + hl * 16), *(const v8h*)(qb + hl * 16 + 8));

  // Scores over 48 key slots (3 x 16-key B tiles, K row-major = dim contiguous)
  v8f sc[3];
  #pragma unroll
  for (int t = 0; t < 3; ++t) {
    const _Float16* kb = K + ((long)b * 2080 + s0 + t * 16 + m) * 256 + h * 32 + hl * 16;
    v16h bk = *(const v16h*)kb;
    v8f z = {};
    sc[t] = WMMA_F32_F16(aq, bk, z);
  }

  const float SCALE = 0.17677669529663687f;          // 1/sqrt(32)
  #pragma unroll
  for (int v = 0; v < 8; ++v) {
    const int q = v + hl * 8;                        // query row within tile
    float mx = -1e30f;
    float sv[3];
    bool  ok[3];
    #pragma unroll
    for (int t = 0; t < 3; ++t) {
      int rel = t * 16 + m - q;                      // key offset within window
      ok[t] = (rel >= 0) && (rel <= 30);
      sv[t] = sc[t][v];
      if (ok[t]) mx = fmaxf(mx, sv[t]);
    }
    #pragma unroll
    for (int o = 1; o < 16; o <<= 1) mx = fmaxf(mx, __shfl_xor(mx, o, 32));
    float e[3];
    float sum = 0.f;
    #pragma unroll
    for (int t = 0; t < 3; ++t) {
      e[t] = ok[t] ? __expf((sv[t] - mx) * SCALE) : 0.f;
      sum += e[t];
    }
    #pragma unroll
    for (int o = 1; o < 16; o <<= 1) sum += __shfl_xor(sum, o, 32);
    float r = 1.0f / sum;
    #pragma unroll
    for (int t = 0; t < 3; ++t) P[w][q][t * 16 + m] = (_Float16)(e[t] * r);
  }
  __syncthreads();                                   // cross-lane LDS communication

  // P as A operand: chunk0 = keys 0..31, chunk1 = keys 32..47 (zero padded)
  v8h zero8 = {};
  v16h a0 = cat16(*(const v8h*)&P[w][m][hl * 16], *(const v8h*)&P[w][m][hl * 16 + 8]);
  v16h a1 = hl ? cat16(zero8, zero8) : cat16(*(const v8h*)&P[w][m][32], *(const v8h*)&P[w][m][40]);

  #pragma unroll
  for (int nt = 0; nt < 2; ++nt) {
    const int dim = h * 32 + nt * 16 + m;
    const _Float16* vb = Vt + ((long)b * 256 + dim) * 2112 + s0;
    v16h b0 = *(const v16h*)(vb + hl * 16);
    v16h b1 = *(const v16h*)(vb + 32 + hl * 16);
    v8f o = {};
    o = WMMA_F32_F16(a0, b0, o);
    o = WMMA_F32_F16(a1, b1, o);
    #pragma unroll
    for (int v = 0; v < 8; ++v)
      Ot[w][v + hl * 8][nt * 16 + m] = (_Float16)o[v];
  }
  __syncthreads();

  // Packed output store: each lane writes one 32B half-row
  {
    int row = ln >> 1, ch = ln & 1;
    long grow = (long)b * 2048 + s0 + row;
    *(v16h*)&O[grow * 256 + h * 32 + ch * 16] = *(const v16h*)&Ot[w][row][ch * 16];
  }
}

// ---------------------------------------------------------------------------
extern "C" void kernel_launch(void* const* d_in, const int* in_sizes, int n_in,
                              void* d_out, int out_size, void* d_ws, size_t ws_size,
                              hipStream_t stream) {
  const int B = 4, S = 2048, H = 8;
  const int SP = 2080;   // padded key rows (2078 rounded up to tile multiple)
  const int VT = 2112;   // Vt key stride (covers attention over-read region)
  const float* eeg = (const float*)d_in[0];
  const float* pps = (const float*)d_in[1];

  char* p = (char*)d_ws;
  auto take = [&](size_t bytes) -> char* {
    char* r = p;
    p += (bytes + 255) & ~(size_t)255;
    return r;
  };

  _Float16* W16[2][4];
  for (int d = 0; d < 2; ++d)
    for (int j = 0; j < 4; ++j) W16[d][j] = (_Float16*)take(65536 * 2);
  _Float16 *Qb[2], *Kb[2], *Vb[2], *Ob[2];
  for (int d = 0; d < 2; ++d) {
    Qb[d] = (_Float16*)take((size_t)B * S * 256 * 2);
    Kb[d] = (_Float16*)take((size_t)B * SP * 256 * 2);
    Vb[d] = (_Float16*)take((size_t)B * 256 * VT * 2);
    Ob[d] = (_Float16*)take((size_t)B * S * 256 * 2);
  }

  // Weight conversion (fp32 -> f16 B-operand layout)
  for (int d = 0; d < 2; ++d) {
    const int base = 2 + d * 8;
    conv_qkv_w<<<256, 256, 0, stream>>>((const float*)d_in[base + 0], W16[d][0]); // Wq
    conv_qkv_w<<<256, 256, 0, stream>>>((const float*)d_in[base + 2], W16[d][1]); // Wk
    conv_qkv_w<<<256, 256, 0, stream>>>((const float*)d_in[base + 4], W16[d][2]); // Wv
    conv_sq_w <<<256, 256, 0, stream>>>((const float*)d_in[base + 6], W16[d][3]); // Wo
  }

  for (int d = 0; d < 2; ++d) {
    const int base = 2 + d * 8;
    const float* q  = d ? pps : eeg;   // d=0: e2p (q=eeg, kv=pps)
    const float* kv = d ? eeg : pps;
    const float* bq = (const float*)d_in[base + 1];
    const float* bk = (const float*)d_in[base + 3];
    const float* bv = (const float*)d_in[base + 5];
    const float* bo = (const float*)d_in[base + 7];

    // Q projection: all B*S rows contiguous, no padding
    proj16<<<dim3(512, 1), 256, 0, stream>>>(q, 0, 0L, W16[d][0], bq,
                                             Qb[d], 0, 0L, 0, B * S, 0);
    // K projection: per-batch, zero-padded rows -> bias only
    proj16<<<dim3(130, B), 256, 0, stream>>>(kv, 0, (long)S * 256, W16[d][1], bk,
                                             Kb[d], 0, (long)SP * 256, 0, S, 15);
    // V projection: per-batch, transposed (dim-major) for attention B operand
    proj16<<<dim3(132, B), 256, 0, stream>>>(kv, 0, (long)S * 256, W16[d][2], bv,
                                             Vb[d], 1, (long)256 * VT, VT, S, 15);
    // Sliding-window attention
    attn31<<<dim3(S / 64, H, B), 128, 0, stream>>>(Qb[d], Kb[d], Vb[d], Ob[d]);
    // Output projection -> fp32 into d_out (fused_eeg then fused_pps)
    //   f16 input path: async DMA global->LDS staging
    float* out = (float*)d_out + (size_t)d * B * S * 256;
    proj16<<<dim3(512, 1), 256, 0, stream>>>(Ob[d], 1, 0L, W16[d][3], bo,
                                             out, 2, 0L, 0, B * S, 0);
  }
}
